// StyleGAN2Generator_13314398618225
// MI455X (gfx1250) — matrix operand order repaired and checked by reading, not compiled, not run on hardware
//
#include <hip/hip_runtime.h>
#include <hip/hip_bf16.h>
#include <math.h>

// ---------------------------------------------------------------------------
// StyleGAN2 generator forward for gfx1250 (MI455X).
// Convs = implicit-GEMM on v_wmma_f32_16x16x32_bf16.
// Block = 128 threads (4 waves) computing a 64-oc x 64-pixel macro-tile:
// one B fragment (NHWC bf16, 2x global_load_b128) feeds 4 WMMAs against a
// 64 x IC weight panel staged in LDS (async global->LDS when available).
// ---------------------------------------------------------------------------

#define BATCHN 8
#define ZDIM   512
#define WDIM   512

#if __has_builtin(__builtin_amdgcn_global_load_async_to_lds_b128)
#define USE_ASYNC_LDS 1
#else
#define USE_ASYNC_LDS 0
#endif

typedef __attribute__((ext_vector_type(16))) __bf16 v16bf;
typedef __attribute__((ext_vector_type(8)))  __bf16 v8bf;
typedef __attribute__((ext_vector_type(8)))  float  v8f;
typedef __attribute__((ext_vector_type(4)))  int    v4i;

__device__ __forceinline__ float leakyf(float v) { return v >= 0.f ? v : 0.2f * v; }

__device__ __forceinline__ void async_drain() {
#if USE_ASYNC_LDS
#if __has_builtin(__builtin_amdgcn_s_wait_asynccnt)
    __builtin_amdgcn_s_wait_asynccnt(0);
#else
    asm volatile("s_wait_asynccnt 0" ::: "memory");
#endif
#endif
}

// ---------------- mapping network: out[b,o] = leaky(in[b,:]·W[o,:] + bias[o])
__global__ void k_map(const float* __restrict__ in, const float* __restrict__ Wm,
                      const float* __restrict__ bias, float* __restrict__ out,
                      int in_d) {
    int idx = blockIdx.x * blockDim.x + threadIdx.x;
    if (idx >= BATCHN * WDIM) return;
    int b = idx / WDIM, o = idx % WDIM;
    const float* xi = in + (size_t)b * in_d;
    const float* wr = Wm + (size_t)o * in_d;
    float acc = 0.f;
    for (int j = 0; j < in_d; ++j) acc += xi[j] * wr[j];
    out[idx] = leakyf(acc + bias[o]);
}

// ---------------- broadcast the learned constant over the batch (NCHW f32)
__global__ void k_bcast_const(const float* __restrict__ c, float* __restrict__ x,
                              int n_per_b) {
    int idx = blockIdx.x * blockDim.x + threadIdx.x;
    if (idx >= BATCHN * n_per_b) return;
    x[idx] = c[idx % n_per_b];
}

// ---------------- style projection: s[b,i] = wlat[b,:]·sp_w[i,:] + sp_b[i]
__global__ void k_style(const float* __restrict__ wlat, const float* __restrict__ spw,
                        const float* __restrict__ spb, float* __restrict__ s, int IC) {
    int idx = blockIdx.x * blockDim.x + threadIdx.x;
    if (idx >= BATCHN * IC) return;
    int b = idx / IC, i = idx % IC;
    const float* wl = wlat + (size_t)b * WDIM;
    const float* wr = spw + (size_t)i * WDIM;
    float acc = 0.f;
    for (int j = 0; j < WDIM; ++j) acc += wl[j] * wr[j];
    s[idx] = acc + spb[i];
}

// ---------------- w2[o,i] = sum_t w[o,i,t]^2
__global__ void k_w2(const float* __restrict__ w, float* __restrict__ w2,
                     int OC, int IC, int taps) {
    int idx = blockIdx.x * blockDim.x + threadIdx.x;
    if (idx >= OC * IC) return;
    const float* p = w + (size_t)idx * taps;
    float acc = 0.f;
    for (int t = 0; t < taps; ++t) acc += p[t] * p[t];
    w2[idx] = acc;
}

// ---------------- demod[b,o] = rsqrt(sum_i s[b,i]^2 * w2[o,i] + 1e-8)
__global__ void k_demod(const float* __restrict__ s, const float* __restrict__ w2,
                        float* __restrict__ d, int OC, int IC) {
    int idx = blockIdx.x * blockDim.x + threadIdx.x;
    if (idx >= BATCHN * OC) return;
    int b = idx / OC, o = idx % OC;
    const float* sb = s + (size_t)b * IC;
    const float* wr = w2 + (size_t)o * IC;
    float acc = 0.f;
    for (int i = 0; i < IC; ++i) acc += sb[i] * sb[i] * wr[i];
    d[idx] = rsqrtf(acc + 1e-8f);
}

// ---------------- weights OIHW -> bf16 A-matrix [OCp64, Kp], K = t*IC + i
// (tap-major so a 32-K block is 32 contiguous input channels of one tap;
//  OC zero-padded to a multiple of 64 so conv blocks are branch-free)
__global__ void k_prep_w(const float* __restrict__ w, __bf16* __restrict__ wb,
                         int OC, int IC, int taps, int Kp, int total) {
    int idx = blockIdx.x * blockDim.x + threadIdx.x;
    if (idx >= total) return;
    int o = idx / Kp, r = idx % Kp;
    int t = r / IC, i = r % IC;
    float v = (o < OC) ? w[((size_t)o * IC + i) * taps + t] : 0.f;
    wb[idx] = (__bf16)v;
}

// ---------------- modulate (and optionally bilinear-2x upsample):
// NCHW f32 in -> NHWC bf16 out. Channel-fastest writes are coalesced.
__global__ void k_mod(const float* __restrict__ x, const float* __restrict__ s,
                      __bf16* __restrict__ xb, int IC, int Hin, int Win,
                      int Hout, int Wout, int up, long long total) {
    long long idx = (long long)blockIdx.x * blockDim.x + threadIdx.x;
    if (idx >= total) return;
    int i = (int)(idx % IC);
    long long pix = idx / IC;
    int wo = (int)(pix % Wout);
    int ho = (int)((pix / Wout) % Hout);
    int b  = (int)(pix / ((long long)Wout * Hout));
    float v;
    if (!up) {
        v = x[(((size_t)b * IC + i) * Hin + ho) * Win + wo];
    } else {
        // resize(scale=2, bilinear, half-pixel centers), edge clamp
        float fy = ho * 0.5f - 0.25f;
        float fx = wo * 0.5f - 0.25f;
        int y0 = (int)floorf(fy), x0 = (int)floorf(fx);
        float wy = fy - (float)y0, wx = fx - (float)x0;
        int y0c = min(max(y0, 0), Hin - 1);
        int y1c = min(max(y0 + 1, 0), Hin - 1);
        int x0c = min(max(x0, 0), Win - 1);
        int x1c = min(max(x0 + 1, 0), Win - 1);
        const float* base = x + ((size_t)b * IC + i) * Hin * Win;
        float v00 = base[y0c * Win + x0c], v01 = base[y0c * Win + x1c];
        float v10 = base[y1c * Win + x0c], v11 = base[y1c * Win + x1c];
        v = (1.f - wy) * ((1.f - wx) * v00 + wx * v01) +
            wy * ((1.f - wx) * v10 + wx * v11);
    }
    v *= s[(size_t)b * IC + i];
    xb[idx] = (__bf16)v;
}

// ---------------- implicit-GEMM modulated conv via v_wmma_f32_16x16x32_bf16
// Grid: (ceil(HW/64), OCp64/64, B). Block 128 threads = 4 waves.
// Wave wv owns pixel tile blockIdx.x*4+wv; all waves cover the same 64 ocs.
// Per tap the 64 x IC bf16 weight panel is staged in LDS (<=64 KB) via
// GLOBAL_LOAD_ASYNC_TO_LDS_B128 (ASYNCcnt) when the builtin exists; per 32-K
// step one B fragment (two 8-channel NHWC runs per lane half) feeds 4
// WMMAs against the 4 A panels.
// A 16x32 lane layout: lane m=l%16, half hi=l>>4 reads runs [k0+hi*8, +7]
// and [k0+16+hi*8, +7]. B 32x16: column n=l%16; lanes 0-15 hold K 0..7,16..23,
// lanes 16-31 hold K 8..15,24..31. C/D: VGPR r -> (M = r + 8*hi, N = l%16).
__global__ void __launch_bounds__(128)
k_conv_wmma(const __bf16* __restrict__ xb, const __bf16* __restrict__ wb,
            const float* __restrict__ demod, const float* __restrict__ bias,
            float* __restrict__ out, int IC, int OC, int H, int W,
            int ks, int pad, int Kp, int act) {
    __shared__ __bf16 lA[64 * 512];
    const int tid  = threadIdx.x;
    const int lane = tid & 31;
    const int wv   = tid >> 5;
    const int nq   = lane & 15;
    const int hi   = lane >> 4;
    const int b    = blockIdx.z;
    const int HW   = H * W;
    const int p    = (blockIdx.x * 4 + wv) * 16 + nq;   // this lane's pixel
    const bool pvalid = p < HW;
    const int pc = pvalid ? p : 0;
    const int ph = pc / W, pw = pc % W;
    const __bf16* xbase = xb + (size_t)b * HW * IC;
    const __bf16* wblk  = wb + (size_t)blockIdx.y * 64 * Kp;

    v8f c[4] = {};
    for (int ky = 0; ky < ks; ++ky) {
        for (int kx = 0; kx < ks; ++kx) {
            const int t = ky * ks + kx;

            // ---- stage this tap's 64 x IC weight panel into LDS
            __syncthreads();   // previous tap's readers are done
            for (int m = wv; m < 64; m += 4) {
                const __bf16* src = wblk + (size_t)m * Kp + t * IC;
                for (int o = lane * 8; o < IC; o += 256) {
#if USE_ASYNC_LDS
                    __builtin_amdgcn_global_load_async_to_lds_b128(
                        (v4i*)(src + o), (v4i*)(lA + m * IC + o), 0, 0);
#else
                    *(v8bf*)(lA + m * IC + o) = *(const v8bf*)(src + o);
#endif
                }
            }
            async_drain();     // s_wait_asynccnt 0 (no-op on fallback path)
            __syncthreads();

            const int hh = ph + ky - pad;
            const int ww = pw + kx - pad;
            const bool ok = pvalid && hh >= 0 && hh < H && ww >= 0 && ww < W;

            for (int i0 = 0; i0 < IC; i0 += 32) {
                // B fragment: channel runs [hi*8, +7] and [16+hi*8, +7] at this lane's pixel
                v16bf bm = {};
                if (ok) {
                    const __bf16* bpix =
                        xbase + (size_t)(hh * W + ww) * IC + i0 + hi * 8;
                    v8bf b0 = *(const v8bf*)(bpix);
                    v8bf b1 = *(const v8bf*)(bpix + 16);
#pragma unroll
                    for (int j = 0; j < 8; ++j) { bm[j] = b0[j]; bm[8 + j] = b1[j]; }
                }

                // 4 A fragments from LDS, 4 WMMAs sharing the B fragment
#pragma unroll
                for (int mt = 0; mt < 4; ++mt) {
                    const __bf16* ar = lA + (mt * 16 + nq) * IC + i0 + hi * 8;
                    v8bf a0 = *(const v8bf*)(ar);
                    v8bf a1 = *(const v8bf*)(ar + 16);
                    v16bf a;
#pragma unroll
                    for (int j = 0; j < 8; ++j) { a[j] = a0[j]; a[8 + j] = a1[j]; }
                    c[mt] = __builtin_amdgcn_wmma_f32_16x16x32_bf16(
                                false, a, false, bm, (short)0, c[mt], false, false);
                }
            }
        }
    }

    // ---- epilogue: demodulate + bias + activation, NCHW f32 out
    if (pvalid) {
#pragma unroll
        for (int mt = 0; mt < 4; ++mt) {
#pragma unroll
            for (int r = 0; r < 8; ++r) {
                int m = blockIdx.y * 64 + mt * 16 + r + hi * 8;
                if (m < OC) {
                    float v = c[mt][r] * demod[(size_t)b * OC + m] + bias[m];
                    v = (act == 0) ? leakyf(v) : tanhf(v);
                    out[((size_t)b * OC + m) * HW + p] = v;
                }
            }
        }
    }
}

// ---------------------------------------------------------------------------
// Host orchestration
// ---------------------------------------------------------------------------
extern "C" void kernel_launch(void* const* d_in, const int* in_sizes, int n_in,
                              void* d_out, int out_size, void* d_ws, size_t ws_size,
                              hipStream_t stream) {
    (void)in_sizes; (void)n_in; (void)out_size; (void)ws_size;

    static const int CH[7] = {512, 512, 512, 512, 512, 256, 128};

    const float* z = (const float*)d_in[0];
    // d_in layout (setup_inputs dict insertion order):
    //  0: z
    //  1+2i / 2+2i : mapping W_i, b_i  (i = 0..7)
    //  17: const [1,512,4,4]
    //  18 + 8*li + {0..3}: layer li conv1 {weight, sp_w, sp_b, bias}
    //  18 + 8*li + {4..7}: layer li conv2 {weight, sp_w, sp_b, bias}
    //  74..77: to_rgb {weight, sp_w, sp_b, bias}
    const float* constp = (const float*)d_in[17];

    // ---- carve workspace
    char* ws = (char*)d_ws;
    size_t off = 0;
    auto carve = [&](size_t bytes) -> void* {
        void* p = ws + off;
        off = (off + bytes + 255) & ~(size_t)255;
        return p;
    };
    float*  wl0   = (float*) carve((size_t)BATCHN * WDIM * 4);
    float*  wl1   = (float*) carve((size_t)BATCHN * WDIM * 4);
    float*  sbuf  = (float*) carve((size_t)BATCHN * 512 * 4);
    float*  dmbuf = (float*) carve((size_t)BATCHN * 512 * 4);
    float*  w2buf = (float*) carve((size_t)512 * 512 * 4);
    __bf16* wbbuf = (__bf16*)carve((size_t)512 * 4640 * 2);
    __bf16* xbbuf = (__bf16*)carve((size_t)BATCHN * 256 * 256 * 256 * 2); // 256 MiB
    float*  actA  = (float*) carve((size_t)BATCHN * 128 * 256 * 256 * 4); // 256 MiB
    float*  actB  = (float*) carve((size_t)BATCHN * 128 * 256 * 256 * 4); // 256 MiB

    // ---- mapping network (8 tiny GEMV layers, ping-pong)
    const float* msrc = z;
    float* wlat = wl1;
    for (int i = 0; i < 8; ++i) {
        float* dst = (i & 1) ? wl1 : wl0;
        k_map<<<(BATCHN * WDIM + 255) / 256, 256, 0, stream>>>(
            msrc, (const float*)d_in[1 + 2 * i], (const float*)d_in[2 + 2 * i],
            dst, (i == 0) ? ZDIM : WDIM);
        msrc = dst;
        wlat = dst;
    }

    // ---- one modulated conv
    auto run_conv = [&](const float* xin, float* xout,
                        const float* w, const float* spw, const float* spb,
                        const float* bias, int IC, int OC, int Hin, int Win,
                        int ks, bool up, int actmode) {
        const int Hout = up ? 2 * Hin : Hin;
        const int Wout = up ? 2 * Win : Win;
        const int taps = ks * ks;
        const int Kp   = taps * IC;          // IC is always a multiple of 32
        const int OCp  = (OC + 63) & ~63;    // pad M to 64 for branch-free blocks
        const int pad  = ks / 2;

        k_style<<<(BATCHN * IC + 255) / 256, 256, 0, stream>>>(wlat, spw, spb, sbuf, IC);
        k_w2<<<(OC * IC + 255) / 256, 256, 0, stream>>>(w, w2buf, OC, IC, taps);
        k_demod<<<(BATCHN * OC + 255) / 256, 256, 0, stream>>>(sbuf, w2buf, dmbuf, OC, IC);
        k_prep_w<<<(OCp * Kp + 255) / 256, 256, 0, stream>>>(w, wbbuf, OC, IC, taps, Kp,
                                                             OCp * Kp);

        const long long nmod = (long long)BATCHN * IC * Hout * Wout;
        k_mod<<<(unsigned)((nmod + 255) / 256), 256, 0, stream>>>(
            xin, sbuf, xbbuf, IC, Hin, Win, Hout, Wout, up ? 1 : 0, nmod);

        const int npix = Hout * Wout;
        dim3 grid((unsigned)((npix + 63) / 64), (unsigned)(OCp / 64), (unsigned)BATCHN);
        k_conv_wmma<<<grid, 128, 0, stream>>>(
            xbbuf, wbbuf, dmbuf, bias, xout, IC, OC, Hout, Wout, ks, pad, Kp, actmode);
    };

    // ---- synthesis
    float* act[2] = {actA, actB};
    int cur = 0;
    k_bcast_const<<<(BATCHN * 512 * 16 + 255) / 256, 256, 0, stream>>>(
        constp, act[cur], 512 * 16);

    int H = 4, W = 4, IC = 512;
    int pidx = 18;
    for (int li = 0; li < 7; ++li) {
        const int OC = CH[li];
        const bool up = (li > 0);
        // conv1
        run_conv(act[cur], act[cur ^ 1],
                 (const float*)d_in[pidx + 0], (const float*)d_in[pidx + 1],
                 (const float*)d_in[pidx + 2], (const float*)d_in[pidx + 3],
                 IC, OC, H, W, 3, up, /*leaky*/0);
        cur ^= 1;
        if (up) { H *= 2; W *= 2; }
        // conv2
        run_conv(act[cur], act[cur ^ 1],
                 (const float*)d_in[pidx + 4], (const float*)d_in[pidx + 5],
                 (const float*)d_in[pidx + 6], (const float*)d_in[pidx + 7],
                 OC, OC, H, W, 3, false, /*leaky*/0);
        cur ^= 1;
        IC = OC;
        pidx += 8;
    }

    // ---- to_rgb (1x1 conv, tanh) straight into d_out
    run_conv(act[cur], (float*)d_out,
             (const float*)d_in[74], (const float*)d_in[75],
             (const float*)d_in[76], (const float*)d_in[77],
             IC, 3, H, W, 1, false, /*tanh*/1);
}
